// LightweightMLPDecoder_84920093376640
// MI455X (gfx1250) — matrix operand order repaired and checked by reading, not compiled, not run on hardware
//
#include <hip/hip_runtime.h>
#include <stdint.h>

typedef __attribute__((ext_vector_type(16))) __bf16 v16bf;
typedef __attribute__((ext_vector_type(8)))  float  v8f;

#define HIDDEN 64
#define DFEAT  128
#define KDIM   256      // 2*DFEAT
#define TILE_M 32       // edges per block iteration (2 x 16-row WMMA subtiles)
#define LDSPITCH 264    // 256 + 8 pad (bf16 elems) -> conflict-free b128 LDS reads
#define BN_EPS 1e-5f

__device__ __forceinline__ unsigned short f32_to_bf16(float f) {
    uint32_t u = __float_as_uint(f);
    u += 0x7FFFu + ((u >> 16) & 1u);   // round-to-nearest-even
    return (unsigned short)(u >> 16);
}

union Frag {
    uint4 u[2];
    v16bf v;
};

// ---------- prologue: bf16 copies of feature tables + W1^T, zero stats ----------
__global__ void k_prep(const float* __restrict__ drug, const float* __restrict__ dis,
                       const float* __restrict__ W1,
                       unsigned short* __restrict__ drug_bf,
                       unsigned short* __restrict__ dis_bf,
                       unsigned short* __restrict__ w1t,
                       float* __restrict__ sums, float* __restrict__ sumsq,
                       int nDrug, int nDis) {
    int tid = blockIdx.x * blockDim.x + threadIdx.x;
    int nthreads = gridDim.x * blockDim.x;
    if (blockIdx.x == 0) {
        if (threadIdx.x < HIDDEN)            sums[threadIdx.x] = 0.f;
        else if (threadIdx.x < 2 * HIDDEN)   sumsq[threadIdx.x - HIDDEN] = 0.f;
    }
    for (int i = tid; i < nDrug; i += nthreads) drug_bf[i] = f32_to_bf16(drug[i]);
    for (int i = tid; i < nDis;  i += nthreads) dis_bf[i]  = f32_to_bf16(dis[i]);
    for (int i = tid; i < KDIM * HIDDEN; i += nthreads) {
        int k = i >> 6, n = i & 63;                 // W1 is [256][64] row-major
        w1t[n * KDIM + k] = f32_to_bf16(W1[i]);     // store as [64][256]
    }
}

// Load this wave's W1^T N-tile as register-resident B fragments.
__device__ __forceinline__ void load_bfrags(const unsigned short* __restrict__ w1t,
                                            int nrow, int half, v16bf* bfr) {
#pragma unroll
    for (int kb = 0; kb < 8; ++kb) {
        Frag f;
        const unsigned short* p = w1t + (size_t)nrow * KDIM + kb * 32 + half * 8;
        f.u[0] = *(const uint4*)(p);
        f.u[1] = *(const uint4*)(p + 16);
        bfr[kb] = f.v;
    }
}

// Cooperative gather of TILE_M edge rows into LDS (bf16, padded pitch).
__device__ __forceinline__ void gather_tile(unsigned short* __restrict__ a_tile,
                                            const int* __restrict__ eidx,
                                            const unsigned short* __restrict__ drug_bf,
                                            const unsigned short* __restrict__ dis_bf,
                                            int tid) {
#pragma unroll
    for (int i = 0; i < 8; ++i) {
        int chunk = tid + i * 128;            // 1024 chunks of 16B (32 rows x 32)
        int row = chunk >> 5;
        int c   = chunk & 31;
        uint4 v = make_uint4(0u, 0u, 0u, 0u);
        if (c < 16) {
            int si = eidx[row];
            if (si >= 0) v = *(const uint4*)(drug_bf + (size_t)si * DFEAT + c * 8);
        } else {
            int di = eidx[32 + row];
            if (di >= 0) v = *(const uint4*)(dis_bf + (size_t)di * DFEAT + (c - 16) * 8);
        }
        *(uint4*)(&a_tile[row * LDSPITCH + c * 8]) = v;
    }
}

// Compute two 16x16 C subtiles: preload ALL A fragments, then run the WMMA chain
// back-to-back so ds_load latency is pipelined instead of serialized per-WMMA.
__device__ __forceinline__ void compute_two_subtiles(const unsigned short* __restrict__ a_tile,
                                                     const v16bf* __restrict__ bfr,
                                                     int l16, int half,
                                                     v8f& c0, v8f& c1) {
    const unsigned short* aRow0 = &a_tile[l16 * LDSPITCH];
    const unsigned short* aRow1 = &a_tile[(16 + l16) * LDSPITCH];
    Frag fa0[8], fa1[8];
#pragma unroll
    for (int kb = 0; kb < 8; ++kb) {
        fa0[kb].u[0] = *(const uint4*)(aRow0 + kb * 32 + half * 8);
        fa0[kb].u[1] = *(const uint4*)(aRow0 + kb * 32 + 16 + half * 8);
        fa1[kb].u[0] = *(const uint4*)(aRow1 + kb * 32 + half * 8);
        fa1[kb].u[1] = *(const uint4*)(aRow1 + kb * 32 + 16 + half * 8);
    }
#pragma unroll
    for (int kb = 0; kb < 8; ++kb) {
        c0 = __builtin_amdgcn_wmma_f32_16x16x32_bf16(
            false, fa0[kb].v, false, bfr[kb], (short)0, c0, false, false);
        c1 = __builtin_amdgcn_wmma_f32_16x16x32_bf16(
            false, fa1[kb].v, false, bfr[kb], (short)0, c1, false, false);
    }
}

__global__ __launch_bounds__(128) void k_pass1(
        const int* __restrict__ src, const int* __restrict__ dst,
        const unsigned short* __restrict__ drug_bf,
        const unsigned short* __restrict__ dis_bf,
        const unsigned short* __restrict__ w1t,
        float* __restrict__ sums, float* __restrict__ sumsq, int E) {
    __shared__ unsigned short a_tile[TILE_M * LDSPITCH];
    __shared__ int eidx[64];
    const int tid  = threadIdx.x;
    const int lane = tid & 31;
    const int wave = tid >> 5;          // 0..3 -> N tile
    const int half = lane >> 4;
    const int l16  = lane & 15;
    const int nrow = wave * 16 + l16;   // hidden channel for this lane

    v16bf bfr[8];
    load_bfrags(w1t, nrow, half, bfr);

    float sAcc = 0.f, qAcc = 0.f;
    const int numTiles = (E + TILE_M - 1) / TILE_M;
    for (int t = blockIdx.x; t < numTiles; t += gridDim.x) {
        if (tid < 64) {
            int m = tid & 31;
            int e = t * TILE_M + m;
            eidx[tid] = (e < E) ? ((tid < 32) ? src[e] : dst[e]) : -1;
        }
        __syncthreads();
        gather_tile(a_tile, eidx, drug_bf, dis_bf, tid);
        __syncthreads();

        v8f c0 = {}, c1 = {};
        compute_two_subtiles(a_tile, bfr, l16, half, c0, c1);
#pragma unroll
        for (int r = 0; r < 8; ++r) {
            sAcc += c0[r] + c1[r];
            qAcc += c0[r] * c0[r] + c1[r] * c1[r];
        }
        __syncthreads();
    }
    // combine the two half-groups (same N channel), then one atomic per channel
    sAcc += __shfl_xor(sAcc, 16, 32);
    qAcc += __shfl_xor(qAcc, 16, 32);
    if (lane < 16) {
        atomicAdd(&sums[wave * 16 + lane], sAcc);
        atomicAdd(&sumsq[wave * 16 + lane], qAcc);
    }
}

// ---------- fold BN into per-channel scale/shift (b1 cancels in training-mode BN) ----------
__global__ void k_stats(const float* __restrict__ sums, const float* __restrict__ sumsq,
                        const float* __restrict__ gamma, const float* __restrict__ beta,
                        float* __restrict__ scale, float* __restrict__ shift, int E) {
    int t = threadIdx.x;
    if (t < HIDDEN) {
        float invE = 1.0f / (float)E;
        float mean = sums[t] * invE;
        float var  = sumsq[t] * invE - mean * mean;
        float inv  = rsqrtf(var + BN_EPS);
        float sc   = gamma[t] * inv;
        scale[t] = sc;
        shift[t] = beta[t] - mean * sc;
    }
}

__global__ __launch_bounds__(128) void k_pass2(
        const int* __restrict__ src, const int* __restrict__ dst,
        const unsigned short* __restrict__ drug_bf,
        const unsigned short* __restrict__ dis_bf,
        const unsigned short* __restrict__ w1t,
        const float* __restrict__ scale, const float* __restrict__ shift,
        const float* __restrict__ W2, const float* __restrict__ b2,
        float* __restrict__ out, int E) {
    __shared__ unsigned short a_tile[TILE_M * LDSPITCH];
    __shared__ int eidx[64];
    __shared__ float part[4][TILE_M];
    const int tid  = threadIdx.x;
    const int lane = tid & 31;
    const int wave = tid >> 5;
    const int half = lane >> 4;
    const int l16  = lane & 15;
    const int nrow = wave * 16 + l16;

    v16bf bfr[8];
    load_bfrags(w1t, nrow, half, bfr);
    const float sc  = scale[nrow];
    const float sh  = shift[nrow];
    const float w2n = W2[nrow];
    const float b2v = b2[0];

    const int numTiles = (E + TILE_M - 1) / TILE_M;
    for (int t = blockIdx.x; t < numTiles; t += gridDim.x) {
        if (tid < 64) {
            int m = tid & 31;
            int e = t * TILE_M + m;
            eidx[tid] = (e < E) ? ((tid < 32) ? src[e] : dst[e]) : -1;
        }
        __syncthreads();
        gather_tile(a_tile, eidx, drug_bf, dis_bf, tid);
        __syncthreads();

        v8f c0 = {}, c1 = {};
        compute_two_subtiles(a_tile, bfr, l16, half, c0, c1);

        // BN + ReLU + dot with W2 slice; reduce over the 16 N-lanes per row
        float p0[8], p1[8];
#pragma unroll
        for (int r = 0; r < 8; ++r) {
            float v0 = fmaxf(fmaf(c0[r], sc, sh), 0.f);
            float v1 = fmaxf(fmaf(c1[r], sc, sh), 0.f);
            p0[r] = v0 * w2n;
            p1[r] = v1 * w2n;
        }
#pragma unroll
        for (int mask = 1; mask <= 8; mask <<= 1) {
#pragma unroll
            for (int r = 0; r < 8; ++r) {
                p0[r] += __shfl_xor(p0[r], mask, 32);
                p1[r] += __shfl_xor(p1[r], mask, 32);
            }
        }
        if (l16 == 0) {
#pragma unroll
            for (int r = 0; r < 8; ++r) {
                part[wave][half * 8 + r]      = p0[r];
                part[wave][16 + half * 8 + r] = p1[r];
            }
        }
        __syncthreads();
        if (tid < TILE_M) {
            int e = t * TILE_M + tid;
            if (e < E)
                out[e] = part[0][tid] + part[1][tid] + part[2][tid] + part[3][tid] + b2v;
        }
        __syncthreads();
    }
}

extern "C" void kernel_launch(void* const* d_in, const int* in_sizes, int n_in,
                              void* d_out, int out_size, void* d_ws, size_t ws_size,
                              hipStream_t stream) {
    (void)n_in; (void)out_size; (void)ws_size;
    const float* drug  = (const float*)d_in[0];
    const float* dis   = (const float*)d_in[1];
    const int*   src   = (const int*)d_in[2];
    const int*   dst   = (const int*)d_in[3];
    const float* W1    = (const float*)d_in[4];
    // d_in[5] = b1: unused — cancels exactly in training-mode BatchNorm
    const float* gamma = (const float*)d_in[6];
    const float* beta  = (const float*)d_in[7];
    const float* W2    = (const float*)d_in[8];
    const float* b2    = (const float*)d_in[9];
    const int E = in_sizes[2];
    const int nDrugElems = in_sizes[0];
    const int nDisElems  = in_sizes[1];

    auto align256 = [](size_t x) { return (x + 255) & ~(size_t)255; };
    char* ws = (char*)d_ws;
    size_t off = 0;
    unsigned short* drug_bf = (unsigned short*)(ws + off); off = align256(off + (size_t)nDrugElems * 2);
    unsigned short* dis_bf  = (unsigned short*)(ws + off); off = align256(off + (size_t)nDisElems * 2);
    unsigned short* w1t     = (unsigned short*)(ws + off); off = align256(off + (size_t)KDIM * HIDDEN * 2);
    float* sums  = (float*)(ws + off); off = align256(off + HIDDEN * 4);
    float* sumsq = (float*)(ws + off); off = align256(off + HIDDEN * 4);
    float* scale = (float*)(ws + off); off = align256(off + HIDDEN * 4);
    float* shift = (float*)(ws + off); off = align256(off + HIDDEN * 4);

    k_prep<<<2048, 256, 0, stream>>>(drug, dis, W1, drug_bf, dis_bf, w1t,
                                     sums, sumsq, nDrugElems, nDisElems);
    k_pass1<<<1024, 128, 0, stream>>>(src, dst, drug_bf, dis_bf, w1t, sums, sumsq, E);
    k_stats<<<1, 64, 0, stream>>>(sums, sumsq, gamma, beta, scale, shift, E);
    k_pass2<<<1024, 128, 0, stream>>>(src, dst, drug_bf, dis_bf, w1t,
                                      scale, shift, W2, b2, (float*)d_out, E);
}
